// Attention_21208548508269
// MI455X (gfx1250) — matrix operand order repaired and checked by reading, not compiled, not run on hardware
//
#include <hip/hip_runtime.h>
#include <hip/hip_bf16.h>

typedef __attribute__((ext_vector_type(16))) _Float16 v16h;
typedef __attribute__((ext_vector_type(8)))  _Float16 v8h;
typedef __attribute__((ext_vector_type(8)))  float    v8f;

// Problem constants (match reference: B=2, T=2048, C=1024, H=16, D=64)
constexpr int AB = 2;
constexpr int AT = 2048;
constexpr int AC = 1024;
constexpr int AH = 16;
constexpr int AD = 64;
constexpr int AM = AB * AT;     // 4096 rows for projections

// ---------------------------------------------------------------------------
// A-fragment loader for v_wmma_f32_16x16x32_f16.
// Arow points at A[m][k0] (row-major, K contiguous). Layout:
//   elements 0..15 : K = hi*16 + e
// ---------------------------------------------------------------------------
__device__ __forceinline__ v16h load_a_frag(const _Float16* Arow, int hi) {
  v16h a = *reinterpret_cast<const v16h*>(Arow + hi * 16);
  return a;
}

// ---------------------------------------------------------------------------
// fp32 -> f16 elementwise convert
// ---------------------------------------------------------------------------
__global__ void f32_to_f16(const float* __restrict__ in, _Float16* __restrict__ out, int n) {
  int i = blockIdx.x * blockDim.x + threadIdx.x;
  if (i < n) out[i] = (_Float16)in[i];
}

// ---------------------------------------------------------------------------
// Register-blocked WMMA GEMM (M=4096, N=K=1024 hardcoded so row offsets fold
// into 24-bit instruction immediates): out[m][n] = sum_k A[m][k]*W[n][k]+bias.
// One wave computes a 32x64 tile: 2 A-frags x 4 B-frags -> 8 v8f accumulators
// (~170 VGPRs peak incl. pipelined loads -> no scratch spills, unlike 4x4).
// mode 0: fp32 [M,N]; mode 1: scatter fp32 into [B,H,T,D].
// ---------------------------------------------------------------------------
__global__ __launch_bounds__(256)
void gemm_wmma(const _Float16* __restrict__ A, const _Float16* __restrict__ W,
               const float* __restrict__ bias, float* __restrict__ out, int mode) {
  constexpr int M = AM, N = AC, K = AC;
  int wid  = threadIdx.x >> 5;
  int lane = threadIdx.x & 31;
  int gw   = blockIdx.x * 8 + wid;
  constexpr int ntn = N >> 6;        // 16 N-tiles of 64
  int mt = gw / ntn;
  int nt = gw % ntn;
  if (mt >= (M >> 5)) return;

  int r  = lane & 15;
  int hi = lane >> 4;

  const _Float16* Abase = A + (size_t)((mt << 5) + r) * K;  // A rows: +i*16*K
  const _Float16* Wbase = W + (size_t)((nt << 6) + r) * K;  // W rows: +j*16*K

  v8f acc[2][4] = {};
  for (int k0 = 0; k0 < K; k0 += 32) {
    v16h a[2], b[4];
#pragma unroll
    for (int i = 0; i < 2; ++i) a[i] = load_a_frag(Abase + i * 16 * K + k0, hi);
#pragma unroll
    for (int j = 0; j < 4; ++j)
      b[j] = *reinterpret_cast<const v16h*>(Wbase + j * 16 * K + k0 + hi * 16);
#pragma unroll
    for (int i = 0; i < 2; ++i)
#pragma unroll
      for (int j = 0; j < 4; ++j)
        acc[i][j] = __builtin_amdgcn_wmma_f32_16x16x32_f16(false, a[i], false, b[j],
                                                           (short)0, acc[i][j],
                                                           false, false);
  }

#pragma unroll
  for (int j = 0; j < 4; ++j) {
    int ncol = (nt << 6) + (j << 4) + r;
    float bv = bias ? bias[ncol] : 0.0f;
    if (mode == 0) {
#pragma unroll
      for (int i = 0; i < 2; ++i) {
#pragma unroll
        for (int v = 0; v < 8; ++v) {
          int mrow = (mt << 5) + (i << 4) + v + 8 * hi;
          out[(size_t)mrow * N + ncol] = acc[i][j][v] + bv;
        }
      }
    } else {
      int h = ncol >> 6;       // AD = 64
      int d = ncol & 63;
#pragma unroll
      for (int i = 0; i < 2; ++i) {
#pragma unroll
        for (int v = 0; v < 8; ++v) {
          int mrow = (mt << 5) + (i << 4) + v + 8 * hi;
          int b_ = mrow / AT;
          int t  = mrow % AT;
          out[(((size_t)(b_ * AH + h)) * AT + t) * AD + d] = acc[i][j][v] + bv;
        }
      }
    }
  }
}

// ---------------------------------------------------------------------------
// RoPE (pairs along D) + convert to f16, layout [B,H,T,D] kept.
// ---------------------------------------------------------------------------
__global__ void rope_f16(const float* __restrict__ in, _Float16* __restrict__ out, int total) {
  int idx = blockIdx.x * blockDim.x + threadIdx.x;
  if (idx >= total) return;
  int dp   = idx % (AD / 2);
  int rest = idx / (AD / 2);
  int t    = rest % AT;
  size_t base = (size_t)rest * AD + 2 * dp;  // rest = bh*T + t
  float inv = powf(10000.0f, -((float)(2 * dp) / (float)AD));
  float ang = (float)t * inv;
  float c = cosf(ang), s = sinf(ang);
  float x1 = in[base], x2 = in[base + 1];
  out[base]     = (_Float16)(x1 * c - x2 * s);
  out[base + 1] = (_Float16)(x1 * s + x2 * c);
}

// ---------------------------------------------------------------------------
// V: fp32 [B,H,T,D] -> f16 transposed [B,H,D,T]
// ---------------------------------------------------------------------------
__global__ void v_trans_f16(const float* __restrict__ in, _Float16* __restrict__ out, int total) {
  int idx = blockIdx.x * blockDim.x + threadIdx.x;
  if (idx >= total) return;
  int d  = idx % AD;
  int t  = (idx / AD) % AT;
  int bh = idx / (AD * AT);
  out[((size_t)bh * AD + d) * AT + t] = (_Float16)in[idx];
}

// ---------------------------------------------------------------------------
// Flash attention (causal, online softmax). One wave owns 16 query rows of
// one (b,h). KV blocks of 32 keys: 4 WMMAs for S, half-wave shfl softmax
// reductions, P transposed C->A layout through wave-private LDS, 4 WMMAs
// for PV. Next-block K/V prefetched (global_prefetch_b8) to hide latency
// behind the softmax VALU work.
// ---------------------------------------------------------------------------
__global__ __launch_bounds__(256)
void flash_attn(const _Float16* __restrict__ Q, const _Float16* __restrict__ K,
                const _Float16* __restrict__ Vt, _Float16* __restrict__ Y) {
  __shared__ alignas(16) _Float16 psh[8][16][32];  // per-wave 16x32 P tile

  int wid  = threadIdx.x >> 5;
  int lane = threadIdx.x & 31;
  int gw   = blockIdx.x * 8 + wid;
  int qtiles = AT >> 4;
  int qt = gw % qtiles;
  int h  = (gw / qtiles) % AH;
  int b  = gw / (qtiles * AH);
  if (b >= AB) return;

  int r  = lane & 15;
  int hi = lane >> 4;
  int qbase = qt << 4;
  size_t bh = (size_t)b * AH + h;
  const _Float16* Qb = Q + bh * AT * AD;
  const _Float16* Kb = K + bh * AT * AD;
  const _Float16* Vb = Vt + bh * (size_t)AD * AT;

  v16h aq0 = load_a_frag(Qb + (size_t)(qbase + r) * AD + 0, hi);
  v16h aq1 = load_a_frag(Qb + (size_t)(qbase + r) * AD + 32, hi);

  float mrow[8], lrow[8];
  v8f acc[4] = {};
#pragma unroll
  for (int v = 0; v < 8; ++v) { mrow[v] = -1e30f; lrow[v] = 0.0f; }

  const float scale = 0.125f;        // 1/sqrt(64)
  int nkv = (qbase + 16 + 31) >> 5;  // kv blocks of 32 covering keys <= qbase+15

  for (int kb = 0; kb < nkv; ++kb) {
    int kk = kb << 5;

    // ---- S = Q K^T for 2 key sub-tiles ----
    const _Float16* k0row = Kb + (size_t)(kk + r) * AD;
    const _Float16* k1row = Kb + (size_t)(kk + 16 + r) * AD;
    v16h b00 = *reinterpret_cast<const v16h*>(k0row + hi * 16);
    v16h b01 = *reinterpret_cast<const v16h*>(k0row + 32 + hi * 16);
    v16h b10 = *reinterpret_cast<const v16h*>(k1row + hi * 16);
    v16h b11 = *reinterpret_cast<const v16h*>(k1row + 32 + hi * 16);
    v8f s0 = {}, s1 = {};
    s0 = __builtin_amdgcn_wmma_f32_16x16x32_f16(false, aq0, false, b00, (short)0, s0, false, false);
    s0 = __builtin_amdgcn_wmma_f32_16x16x32_f16(false, aq1, false, b01, (short)0, s0, false, false);
    s1 = __builtin_amdgcn_wmma_f32_16x16x32_f16(false, aq0, false, b10, (short)0, s1, false, false);
    s1 = __builtin_amdgcn_wmma_f32_16x16x32_f16(false, aq1, false, b11, (short)0, s1, false, false);

    // prefetch next kv block (K rows + V rows) while we do softmax VALU work
    if (kb + 1 < nkv) {
      __builtin_prefetch(Kb + (size_t)(kk + 32 + r) * AD, 0, 1);
      __builtin_prefetch(Kb + (size_t)(kk + 48 + r) * AD, 0, 1);
      __builtin_prefetch(Vb + (size_t)r * AT + kk + 32, 0, 1);
      __builtin_prefetch(Vb + (size_t)(32 + r) * AT + kk + 32, 0, 1);
    }

    // ---- online softmax (row = v + 8*hi lives in one 16-lane half) ----
    float alpha[8];
#pragma unroll
    for (int v = 0; v < 8; ++v) {
      int qi  = qbase + v + 8 * hi;
      int kj0 = kk + r;
      int kj1 = kk + 16 + r;
      float e0 = (kj0 > qi) ? -1e30f : s0[v] * scale;
      float e1 = (kj1 > qi) ? -1e30f : s1[v] * scale;
      float mx = fmaxf(e0, e1);
      for (int off = 1; off < 16; off <<= 1)
        mx = fmaxf(mx, __shfl_xor(mx, off, 32));
      float mnew = fmaxf(mrow[v], mx);
      alpha[v] = __expf(mrow[v] - mnew);
      float p0 = (kj0 > qi) ? 0.0f : __expf(e0 - mnew);
      float p1 = (kj1 > qi) ? 0.0f : __expf(e1 - mnew);
      psh[wid][v + 8 * hi][r]      = (_Float16)p0;
      psh[wid][v + 8 * hi][16 + r] = (_Float16)p1;
      float rs = p0 + p1;
      for (int off = 1; off < 16; off <<= 1)
        rs += __shfl_xor(rs, off, 32);
      lrow[v] = alpha[v] * lrow[v] + rs;
      mrow[v] = mnew;
    }

    // rescale running accumulators
#pragma unroll
    for (int f = 0; f < 4; ++f)
#pragma unroll
      for (int v = 0; v < 8; ++v) acc[f][v] *= alpha[v];

    // ---- P (C layout) -> A fragment via LDS ----
    v8h a0 = *reinterpret_cast<const v8h*>(&psh[wid][r][hi * 16]);
    v8h a1 = *reinterpret_cast<const v8h*>(&psh[wid][r][hi * 16 + 8]);
    v16h ap = {};
#pragma unroll
    for (int i = 0; i < 8; ++i) { ap[i] = a0[i]; ap[i + 8] = a1[i]; }

    // ---- PV: 4 d-tiles of 16, contraction over the 32 keys ----
#pragma unroll
    for (int f = 0; f < 4; ++f) {
      int d = (f << 4) + r;
      v16h bv = *reinterpret_cast<const v16h*>(Vb + (size_t)d * AT + kk + hi * 16);
      acc[f] = __builtin_amdgcn_wmma_f32_16x16x32_f16(false, ap, false, bv, (short)0,
                                                      acc[f], false, false);
    }
  }

  // ---- epilogue: y[b, t, h*64+d] = acc / l ----
#pragma unroll
  for (int f = 0; f < 4; ++f) {
    int d = (f << 4) + r;
    int c = h * AD + d;
#pragma unroll
    for (int v = 0; v < 8; ++v) {
      int t = qbase + v + 8 * hi;
      Y[((size_t)b * AT + t) * AC + c] = (_Float16)(acc[f][v] / lrow[v]);
    }
  }
}

// ---------------------------------------------------------------------------
// Launch
// ---------------------------------------------------------------------------
extern "C" void kernel_launch(void* const* d_in, const int* in_sizes, int n_in,
                              void* d_out, int out_size, void* d_ws, size_t ws_size,
                              hipStream_t stream) {
  (void)in_sizes; (void)n_in; (void)out_size; (void)ws_size;

  const float* x  = (const float*)d_in[0];
  const float* Wq = (const float*)d_in[1];
  const float* bq = (const float*)d_in[2];
  const float* Wk = (const float*)d_in[3];
  const float* bk = (const float*)d_in[4];
  const float* Wv = (const float*)d_in[5];
  const float* bv = (const float*)d_in[6];
  const float* Wo = (const float*)d_in[7];
  const float* bo = (const float*)d_in[8];
  float* out = (float*)d_out;

  // Workspace carve-up (256B aligned)
  char* ws = (char*)d_ws;
  size_t off = 0;
  auto alloc = [&](size_t bytes) {
    void* p = ws + off;
    off += (bytes + 255) & ~(size_t)255;
    return p;
  };
  const size_t nx = (size_t)AM * AC;   // 4M activations
  const size_t nw = (size_t)AC * AC;   // 1M weights
  _Float16* xh  = (_Float16*)alloc(nx * 2);
  _Float16* wqh = (_Float16*)alloc(nw * 2);
  _Float16* wkh = (_Float16*)alloc(nw * 2);
  _Float16* wvh = (_Float16*)alloc(nw * 2);
  _Float16* woh = (_Float16*)alloc(nw * 2);
  float*    Qf  = (float*)alloc(nx * 4);
  float*    Kf  = (float*)alloc(nx * 4);
  float*    Vf  = (float*)alloc(nx * 4);
  _Float16* Qh  = (_Float16*)alloc(nx * 2);
  _Float16* Kh  = (_Float16*)alloc(nx * 2);
  _Float16* Vth = (_Float16*)alloc(nx * 2);
  _Float16* yh  = (_Float16*)alloc(nx * 2);

  const int THREADS = 256;

  // 1) fp32 -> f16 conversions
  f32_to_f16<<<(int)((nx + THREADS - 1) / THREADS), THREADS, 0, stream>>>(x, xh, (int)nx);
  f32_to_f16<<<(int)((nw + THREADS - 1) / THREADS), THREADS, 0, stream>>>(Wq, wqh, (int)nw);
  f32_to_f16<<<(int)((nw + THREADS - 1) / THREADS), THREADS, 0, stream>>>(Wk, wkh, (int)nw);
  f32_to_f16<<<(int)((nw + THREADS - 1) / THREADS), THREADS, 0, stream>>>(Wv, wvh, (int)nw);
  f32_to_f16<<<(int)((nw + THREADS - 1) / THREADS), THREADS, 0, stream>>>(Wo, woh, (int)nw);

  // 2) Q/K/V projections: 32x64 tile per wave -> (4096/32)*(1024/64)=2048 waves
  int tiles = (AM >> 5) * (AC >> 6);           // 2048
  int gblk  = tiles / (THREADS / 32);          // 256 blocks of 8 waves
  gemm_wmma<<<gblk, THREADS, 0, stream>>>(xh, wqh, bq, Qf, 1);
  gemm_wmma<<<gblk, THREADS, 0, stream>>>(xh, wkh, bk, Kf, 1);
  gemm_wmma<<<gblk, THREADS, 0, stream>>>(xh, wvh, bv, Vf, 1);

  // 3) RoPE(Q), RoPE(K) -> f16; V -> f16 transposed [B,H,D,T]
  int pairs = AB * AH * AT * (AD / 2);         // 2M
  rope_f16<<<(pairs + THREADS - 1) / THREADS, THREADS, 0, stream>>>(Qf, Qh, pairs);
  rope_f16<<<(pairs + THREADS - 1) / THREADS, THREADS, 0, stream>>>(Kf, Kh, pairs);
  int velems = AB * AH * AT * AD;              // 4M
  v_trans_f16<<<(velems + THREADS - 1) / THREADS, THREADS, 0, stream>>>(Vf, Vth, velems);

  // 4) Flash attention: B*H*(T/16) = 4096 waves -> 512 blocks
  int fwaves = AB * AH * (AT >> 4);
  flash_attn<<<fwaves / (THREADS / 32), THREADS, 0, stream>>>(Qh, Kh, Vth, yh);

  // 5) Output projection -> fp32 d_out [B,T,C]
  gemm_wmma<<<gblk, THREADS, 0, stream>>>(yh, woh, bo, out, 0);
}
